// SwinClassifier_80212809220266
// MI455X (gfx1250) — compile-verified, matrix operand order checked by reading, not run on hardware
//
#include <hip/hip_runtime.h>
#include <math.h>

// ---------------- types ----------------
typedef __bf16 bf16;
typedef __attribute__((ext_vector_type(16))) __bf16 v16bf;
typedef __attribute__((ext_vector_type(8)))  __bf16 v8bf;
typedef __attribute__((ext_vector_type(8)))  float  v8f;
typedef int v4i __attribute__((vector_size(16)));

#define TOKENS 32768   // B*H*W = 8*64*64
#define DIM    128
#define NWIN   512     // 8 * 64 windows
#define NB     4

#if __has_builtin(__builtin_amdgcn_global_load_async_to_lds_b128)
#define HAVE_ASYNC_LDS 1
#else
#define HAVE_ASYNC_LDS 0
#endif

__device__ __forceinline__ v8f vzero() {
  v8f z;
#pragma unroll
  for (int i = 0; i < 8; ++i) z[i] = 0.0f;
  return z;
}

__device__ __forceinline__ v8f wmma_bf16(v16bf a, v16bf b, v8f c) {
  // v_wmma_f32_16x16x32_bf16: (neg_a, A, neg_b, B, c_mod, C, reuse_a, reuse_b)
  return __builtin_amdgcn_wmma_f32_16x16x32_bf16(false, a, false, b, (short)0, c,
                                                 false, false);
}

// Load a 16x32 bf16 operand fragment from a row-major matrix (row stride ld,
// tile origin (row0, k0)). Works for A and for W stored as [N][K] (B = W^T).
// CDNA5 layout: lanes 0-15 hold rows 0-15 with K={0..7,16..23};
// lanes 16-31 hold rows 0-15 with K={8..15,24..31}.
__device__ __forceinline__ v16bf load_frag(const bf16* p, int ld, int row0, int k0) {
  int lane = threadIdx.x & 31;
  int r = lane & 15, half = lane >> 4;
  const bf16* base = p + (size_t)(row0 + r) * ld + k0 + half * 8;
  v8bf lo = *(const v8bf*)(base);
  v8bf hi = *(const v8bf*)(base + 16);
  v16bf f;
#pragma unroll
  for (int i = 0; i < 8; ++i) { f[i] = lo[i]; f[i + 8] = hi[i]; }
  return f;
}

// Cooperatively stage nElems bf16 (multiple of 8; chunk count uniform across the
// 256-thread block) from global into LDS. Uses gfx1250 async LDS loads when the
// builtin is available (ASYNCcnt path), else a plain copy. Ends with a barrier.
__device__ __forceinline__ void stage_to_lds(const bf16* __restrict__ gsrc,
                                             bf16* lds, int nElems) {
  int chunks = nElems >> 3;  // 16B chunks
  for (int idx = threadIdx.x; idx < chunks; idx += 256) {
#if HAVE_ASYNC_LDS
    __builtin_amdgcn_global_load_async_to_lds_b128(
        (__attribute__((address_space(1))) v4i*)(
            (__attribute__((address_space(1))) void*)(gsrc + idx * 8)),
        (__attribute__((address_space(3))) v4i*)(
            (__attribute__((address_space(3))) void*)(lds + idx * 8)),
        0, 0);
#else
    ((v8bf*)lds)[idx] = ((const v8bf*)gsrc)[idx];
#endif
  }
#if HAVE_ASYNC_LDS
#if __has_builtin(__builtin_amdgcn_s_wait_asynccnt)
  __builtin_amdgcn_s_wait_asynccnt(0);
#else
  asm volatile("s_wait_asynccnt 0" ::: "memory");
#endif
#endif
  __syncthreads();
}

// window-ordered row -> flat token index (handles shifted roll)
__device__ __forceinline__ int row_to_token(int row, int shifted) {
  int win = row >> 6, n = row & 63;
  int b = win >> 6, wi = win & 63;
  int wh = wi >> 3, ww = wi & 7;
  int r = n >> 3, c = n & 7;
  int h = wh * 8 + r, w = ww * 8 + c;
  if (shifted) { h = (h + 4) & 63; w = (w + 4) & 63; }
  return (b << 12) + (h << 6) + w;
}

__device__ __forceinline__ int region64(int p) {
  return (p < 56) ? 0 : ((p < 60) ? 1 : 2);
}

// ---------------- LayerNorm (+cast to bf16, optional window gather) ---------
__global__ void ln_kernel(const float* __restrict__ X, const float* __restrict__ gw,
                          const float* __restrict__ gb, bf16* __restrict__ outp,
                          int windowOrder, int shifted) {
  int row = blockIdx.x * 8 + (threadIdx.x >> 5);
  int lane = threadIdx.x & 31;
  int src = windowOrder ? row_to_token(row, shifted) : row;
  const float* xp = X + (size_t)src * DIM + lane * 4;
  float4 v = *(const float4*)xp;
  float s = v.x + v.y + v.z + v.w;
#pragma unroll
  for (int m = 16; m >= 1; m >>= 1) s += __shfl_xor(s, m, 32);
  float mean = s * (1.0f / 128.0f);
  float dx = v.x - mean, dy = v.y - mean, dz = v.z - mean, dw = v.w - mean;
  float s2 = dx * dx + dy * dy + dz * dz + dw * dw;
#pragma unroll
  for (int m = 16; m >= 1; m >>= 1) s2 += __shfl_xor(s2, m, 32);
  float rstd = rsqrtf(s2 * (1.0f / 128.0f) + 1e-5f);
  int c = lane * 4;
  bf16* op = outp + (size_t)row * DIM + c;
  op[0] = (bf16)(dx * rstd * gw[c + 0] + gb[c + 0]);
  op[1] = (bf16)(dy * rstd * gw[c + 1] + gb[c + 1]);
  op[2] = (bf16)(dz * rstd * gw[c + 2] + gb[c + 2]);
  op[3] = (bf16)(dw * rstd * gw[c + 3] + gb[c + 3]);
}

// ---------------- generic WMMA GEMM: C[M,N] = A[M,K] * W[N,K]^T + bias -------
// Block: 256 threads (8 waves), each wave a 16x64 strip; the 64xK weight strip
// is staged once into LDS (async when available) and shared by all 8 waves.
// EPI 0: store bf16   1: GELU(exact erf) then store bf16
// EPI 2: scatter-add fp32 at remapped token row (proj residual)
// EPI 3: add fp32 at identity row (fc2 residual)
template <int EPI>
__global__ void gemm_wmma_kernel(const bf16* __restrict__ A, const bf16* __restrict__ W,
                                 const float* __restrict__ bias, void* __restrict__ Out,
                                 int K, int ldo, int shifted) {
  extern __shared__ char smem[];
  bf16* Wl = (bf16*)smem;  // 64 x K bf16
  int wave = threadIdx.x >> 5, lane = threadIdx.x & 31;
  int row0 = (blockIdx.x * 8 + wave) * 16;
  int n0 = blockIdx.y * 64;

  stage_to_lds(W + (size_t)n0 * K, Wl, 64 * K);

  v8f acc[4];
#pragma unroll
  for (int j = 0; j < 4; ++j) acc[j] = vzero();
  for (int k0 = 0; k0 < K; k0 += 32) {
    v16bf a = load_frag(A, K, row0, k0);
#pragma unroll
    for (int j = 0; j < 4; ++j)
      acc[j] = wmma_bf16(a, load_frag(Wl, K, 16 * j, k0), acc[j]);
  }
  int rr = lane & 15, half = lane >> 4;
#pragma unroll
  for (int j = 0; j < 4; ++j) {
    int col = n0 + 16 * j + rr;
    float bs = bias ? bias[col] : 0.0f;
#pragma unroll
    for (int e = 0; e < 8; ++e) {
      int row = row0 + half * 8 + e;
      float v = acc[j][e] + bs;
      if (EPI == 1) v = 0.5f * v * (1.0f + erff(v * 0.70710678f));
      if (EPI <= 1) {
        ((bf16*)Out)[(size_t)row * ldo + col] = (bf16)v;
      } else if (EPI == 2) {
        int t = row_to_token(row, shifted);
        ((float*)Out)[(size_t)t * ldo + col] += v;
      } else {
        ((float*)Out)[(size_t)row * ldo + col] += v;
      }
    }
  }
}

// ---------------- repack V into [win][head][d=32][m=64] bf16 -----------------
__global__ void repack_v_kernel(const bf16* __restrict__ qkv, bf16* __restrict__ vT) {
  int idx = blockIdx.x * 256 + threadIdx.x;  // 512*4*32*64 total
  int m = idx & 63;
  int d = (idx >> 6) & 31;
  int hd = (idx >> 11) & 3;
  int win = idx >> 13;
  vT[idx] = qkv[((size_t)win * 64 + m) * 384 + 256 + hd * 32 + d];
}

// ---------------- window attention: one wave per (window, head) --------------
__global__ void attn_kernel(const bf16* __restrict__ qkv, const bf16* __restrict__ vT,
                            const float* __restrict__ rpb, bf16* __restrict__ obuf,
                            int shifted) {
  extern __shared__ char smem[];
  float* S = (float*)smem;                                  // 64*64 f32
  bf16* P = (bf16*)(smem + 64 * 64 * 4);                    // 64*64 bf16
  float* rb = (float*)(smem + 64 * 64 * 4 + 64 * 64 * 2);   // 225 f32
  int lane = threadIdx.x & 31;
  int win = blockIdx.x, head = blockIdx.y;
  for (int i = threadIdx.x; i < 225; i += 32) rb[i] = rpb[i * 4 + head];
  __syncthreads();

  int wi = win & 63, wh = wi >> 3, ww = wi & 7;
  const bf16* qb = qkv + (size_t)win * 64 * 384 + head * 32;  // q cols
  const bf16* kb = qb + 128;                                  // k cols
  int rr = lane & 15, half = lane >> 4;
  float mshift = shifted ? -100.0f : 0.0f;  // branch-free shift mask

  // S = q k^T + bias (+ shift mask)
  v16bf Bf[4];
#pragma unroll
  for (int j = 0; j < 4; ++j) Bf[j] = load_frag(kb, 384, 16 * j, 0);
#pragma unroll
  for (int i = 0; i < 4; ++i) {
    v16bf Af = load_frag(qb, 384, 16 * i, 0);
#pragma unroll
    for (int j = 0; j < 4; ++j) {
      v8f acc = vzero();
      acc = wmma_bf16(Af, Bf[j], acc);
      int nI = 16 * j + rr;
      int r2 = nI >> 3, c2 = nI & 7;
      int labn = region64(wh * 8 + r2) * 3 + region64(ww * 8 + c2);
#pragma unroll
      for (int e = 0; e < 8; ++e) {
        int mI = 16 * i + half * 8 + e;
        int r1 = mI >> 3, c1 = mI & 7;
        int labm = region64(wh * 8 + r1) * 3 + region64(ww * 8 + c1);
        float bias = rb[(r1 - r2 + 7) * 15 + (c1 - c2 + 7)];
        float mv = (labm != labn) ? mshift : 0.0f;
        S[mI * 64 + nI] = acc[e] + bias + mv;
      }
    }
  }

  // row softmax -> P (bf16); cache exp() back into S (one transcendental/elem)
#pragma unroll
  for (int t = 0; t < 2; ++t) {
    int row = lane + 32 * t;
    float mx = -1e30f;
    for (int n = 0; n < 64; ++n) mx = fmaxf(mx, S[row * 64 + n]);
    float sm = 0.0f;
    for (int n = 0; n < 64; ++n) {
      float e = __expf(S[row * 64 + n] - mx);
      S[row * 64 + n] = e;
      sm += e;
    }
    float inv = 1.0f / sm;
    for (int n = 0; n < 64; ++n) P[row * 64 + n] = (bf16)(S[row * 64 + n] * inv);
  }

  // O = P * V  (vT is [d=32][m=64] row-major => B-operand rows contiguous)
  const bf16* vb = vT + (size_t)(win * 4 + head) * 32 * 64;
#pragma unroll
  for (int i = 0; i < 4; ++i) {
    v16bf A0 = load_frag(P, 64, 16 * i, 0);
    v16bf A1 = load_frag(P, 64, 16 * i, 32);
#pragma unroll
    for (int jn = 0; jn < 2; ++jn) {
      v8f acc = vzero();
      acc = wmma_bf16(A0, load_frag(vb, 64, 16 * jn, 0), acc);
      acc = wmma_bf16(A1, load_frag(vb, 64, 16 * jn, 32), acc);
      int col = head * 32 + 16 * jn + rr;
#pragma unroll
      for (int e = 0; e < 8; ++e) {
        int row = win * 64 + 16 * i + half * 8 + e;
        obuf[(size_t)row * DIM + col] = (bf16)acc[e];
      }
    }
  }
}

// ------- fused expand GEMM + LayerNorm(2048) + pixel-shuffle head + softmax --
__global__ void expand_head_kernel(const bf16* __restrict__ xbf,
                                   const bf16* __restrict__ expw,
                                   const float* __restrict__ gw,
                                   const float* __restrict__ gb,
                                   const float* __restrict__ headw,
                                   float* __restrict__ out) {
  extern __shared__ char smem[];
  float* Y = (float*)smem;                          // 16 x 2048 f32
  float* red = (float*)(smem + 16 * 2048 * 4);      // 16 x 16
  float* red2 = red + 256;                          // 16 x 16
  bf16* Xl = (bf16*)(smem + 16 * 2048 * 4 + 2 * 256 * 4);  // 16 x 128 bf16
  int wave = threadIdx.x >> 5, lane = threadIdx.x & 31;
  int rr = lane & 15, half = lane >> 4;
  int t0 = blockIdx.x * 16;

  // stage this block's 16x128 activation tile once (async LDS path)
  stage_to_lds(xbf + (size_t)t0 * DIM, Xl, 16 * DIM);

  // Y[16,2048] = Xl[16,128] @ expand_w[2048,128]^T  (no bias)
  for (int nt = wave; nt < 128; nt += 8) {
    v8f acc = vzero();
#pragma unroll
    for (int k0 = 0; k0 < 128; k0 += 32)
      acc = wmma_bf16(load_frag(Xl, 128, 0, k0),
                      load_frag(expw, 128, nt * 16, k0), acc);
#pragma unroll
    for (int e = 0; e < 8; ++e) Y[(half * 8 + e) * 2048 + nt * 16 + rr] = acc[e];
  }
  __syncthreads();

  // LayerNorm over 2048 channels per token (16 threads cooperate per token)
  int token = threadIdx.x >> 4, part = threadIdx.x & 15;
  int c0 = part * 128;
  float s = 0.0f, s2 = 0.0f;
  for (int c = c0; c < c0 + 128; ++c) {
    float v = Y[token * 2048 + c];
    s += v; s2 += v * v;
  }
  red[token * 16 + part] = s;
  red2[token * 16 + part] = s2;
  __syncthreads();
  float ts = 0.0f, ts2 = 0.0f;
  for (int p = 0; p < 16; ++p) { ts += red[token * 16 + p]; ts2 += red2[token * 16 + p]; }
  float mean = ts * (1.0f / 2048.0f);
  float var = ts2 * (1.0f / 2048.0f) - mean * mean;
  float rstd = rsqrtf(var + 1e-5f);
  for (int c = c0; c < c0 + 128; ++c)
    Y[token * 2048 + c] = (Y[token * 2048 + c] - mean) * rstd * gw[c] + gb[c];
  __syncthreads();

  // pixel shuffle + 10-class head + softmax; thread = (token, subpixel)
  float lg[10];
#pragma unroll
  for (int k = 0; k < 10; ++k) {
    float a = 0.0f;
    for (int d = 0; d < 128; ++d)
      a += Y[token * 2048 + d * 16 + part] * headw[k * 128 + d];
    lg[k] = a;
  }
  float mx = lg[0];
#pragma unroll
  for (int k = 1; k < 10; ++k) mx = fmaxf(mx, lg[k]);
  float sm = 0.0f;
#pragma unroll
  for (int k = 0; k < 10; ++k) { lg[k] = __expf(lg[k] - mx); sm += lg[k]; }
  float inv = 1.0f / sm;
  int t = t0 + token, b = t >> 12, h = (t >> 6) & 63, w = t & 63;
  int i = part >> 2, j = part & 3;
  int h4 = h * 4 + i, w4 = w * 4 + j;
#pragma unroll
  for (int k = 0; k < 10; ++k)
    out[(((size_t)b * 10 + k) * 256 + h4) * 256 + w4] = lg[k] * inv;
}

// ---------------- small utility kernels --------------------------------------
__global__ void cast_bf16_kernel(const float* __restrict__ src, bf16* __restrict__ dst,
                                 int n, float scale, int scaleCount) {
  int i = blockIdx.x * 256 + threadIdx.x;
  if (i < n) {
    float v = src[i];
    if (i < scaleCount) v *= scale;
    dst[i] = (bf16)v;
  }
}

__global__ void scale_bias_kernel(const float* __restrict__ src, float* __restrict__ dst,
                                  int n, float scale, int scaleCount) {
  int i = blockIdx.x * 256 + threadIdx.x;
  if (i < n) {
    float v = src[i];
    if (i < scaleCount) v *= scale;
    dst[i] = v;
  }
}

// ---------------- host launcher ----------------------------------------------
extern "C" void kernel_launch(void* const* d_in, const int* in_sizes, int n_in,
                              void* d_out, int out_size, void* d_ws, size_t ws_size,
                              hipStream_t stream) {
  const float* x_in = (const float*)d_in[0];
  const float* ln1_w = (const float*)d_in[1];
  const float* ln1_b = (const float*)d_in[2];
  const float* qkv_w = (const float*)d_in[3];
  const float* qkv_b = (const float*)d_in[4];
  const float* proj_w = (const float*)d_in[5];
  const float* proj_b = (const float*)d_in[6];
  const float* rpb = (const float*)d_in[7];
  const float* ln2_w = (const float*)d_in[8];
  const float* ln2_b = (const float*)d_in[9];
  const float* fc1_w = (const float*)d_in[10];
  const float* fc1_b = (const float*)d_in[11];
  const float* fc2_w = (const float*)d_in[12];
  const float* fc2_b = (const float*)d_in[13];
  const float* expand_w = (const float*)d_in[14];
  const float* exnorm_w = (const float*)d_in[15];
  const float* exnorm_b = (const float*)d_in[16];
  const float* head_w = (const float*)d_in[17];

  char* ws = (char*)d_ws;
  size_t off = 0;
  auto alloc = [&](size_t bytes) -> char* {
    char* p = ws + off;
    off = (off + bytes + 255) & ~(size_t)255;
    return p;
  };

  float* xres = (float*)alloc((size_t)TOKENS * DIM * 4);       // residual stream
  bf16* hbuf = (bf16*)alloc((size_t)TOKENS * DIM * 2);         // LN output
  bf16* qkvbuf = (bf16*)alloc((size_t)TOKENS * 384 * 2);       // qkv output
  bf16* vT = (bf16*)alloc((size_t)NWIN * 4 * 32 * 64 * 2);     // transposed V
  bf16* obuf = (bf16*)alloc((size_t)TOKENS * DIM * 2);         // attn output
  bf16* mbuf = (bf16*)alloc((size_t)TOKENS * 512 * 2);         // MLP hidden
  bf16* xbf = (bf16*)alloc((size_t)TOKENS * DIM * 2);          // pre-expand cast
  bf16* wq = (bf16*)alloc(384 * 128 * 2);
  bf16* wp = (bf16*)alloc(128 * 128 * 2);
  bf16* w1 = (bf16*)alloc(512 * 128 * 2);
  bf16* w2 = (bf16*)alloc(128 * 512 * 2);
  bf16* we = (bf16*)alloc(2048 * 128 * 2);
  float* qb_s = (float*)alloc(384 * 4);

  (void)hipMemcpyAsync(xres, x_in, (size_t)TOKENS * DIM * 4,
                       hipMemcpyDeviceToDevice, stream);

  const float qscale = 0.17677669529663687f;  // HD^-0.5, HD=32
  const size_t attn_lds = 64 * 64 * 4 + 64 * 64 * 2 + 256 * 4;       // S + P + rpb
  const size_t exp_lds = 16 * 2048 * 4 + 2 * 256 * 4 + 16 * 128 * 2; // Y + red + Xl

  for (int blk = 0; blk < NB; ++blk) {
    int shifted = blk & 1;
    // weight casts (q scale folded into first 128 rows of qkv_w and qkv_b)
    cast_bf16_kernel<<<(384 * 128 + 255) / 256, 256, 0, stream>>>(
        qkv_w + (size_t)blk * 384 * 128, wq, 384 * 128, qscale, 128 * 128);
    scale_bias_kernel<<<2, 256, 0, stream>>>(qkv_b + blk * 384, qb_s, 384, qscale, 128);
    cast_bf16_kernel<<<64, 256, 0, stream>>>(proj_w + (size_t)blk * 128 * 128, wp,
                                             128 * 128, 1.0f, 0);
    cast_bf16_kernel<<<256, 256, 0, stream>>>(fc1_w + (size_t)blk * 512 * 128, w1,
                                              512 * 128, 1.0f, 0);
    cast_bf16_kernel<<<256, 256, 0, stream>>>(fc2_w + (size_t)blk * 128 * 512, w2,
                                              128 * 512, 1.0f, 0);

    // LN1 with window(+shift) gather -> hbuf (window-ordered rows)
    ln_kernel<<<TOKENS / 8, 256, 0, stream>>>(xres, ln1_w + blk * 128,
                                              ln1_b + blk * 128, hbuf, 1, shifted);
    // qkv GEMM: [32768,128] x [384,128]^T   (LDS-staged weights: 16KB)
    gemm_wmma_kernel<0><<<dim3(TOKENS / 128, 6), 256, 64 * 128 * 2, stream>>>(
        hbuf, wq, qb_s, qkvbuf, 128, 384, 0);
    // transpose V
    repack_v_kernel<<<(NWIN * 4 * 32 * 64) / 256, 256, 0, stream>>>(qkvbuf, vT);
    // attention per (window, head)
    attn_kernel<<<dim3(NWIN, 4), 32, attn_lds, stream>>>(
        qkvbuf, vT, rpb + (size_t)blk * 225 * 4, obuf, shifted);
    // proj GEMM, scatter-add residual back to token order
    gemm_wmma_kernel<2><<<dim3(TOKENS / 128, 2), 256, 64 * 128 * 2, stream>>>(
        obuf, wp, proj_b + blk * 128, xres, 128, 128, shifted);

    // LN2 (token order) -> hbuf
    ln_kernel<<<TOKENS / 8, 256, 0, stream>>>(xres, ln2_w + blk * 128,
                                              ln2_b + blk * 128, hbuf, 0, 0);
    // fc1 + GELU (LDS-staged weights: 16KB)
    gemm_wmma_kernel<1><<<dim3(TOKENS / 128, 8), 256, 64 * 128 * 2, stream>>>(
        hbuf, w1, fc1_b + blk * 512, mbuf, 128, 512, 0);
    // fc2, add residual (LDS-staged weights: 64KB)
    gemm_wmma_kernel<3><<<dim3(TOKENS / 128, 2), 256, 64 * 512 * 2, stream>>>(
        mbuf, w2, fc2_b + blk * 128, xres, 512, 128, 0);
  }

  // final: cast x, cast expand weights, fused expand+LN+head+softmax
  cast_bf16_kernel<<<(TOKENS * DIM) / 256, 256, 0, stream>>>(xres, xbf,
                                                             TOKENS * DIM, 1.0f, 0);
  cast_bf16_kernel<<<(2048 * 128 + 255) / 256, 256, 0, stream>>>(expand_w, we,
                                                                 2048 * 128, 1.0f, 0);
  expand_head_kernel<<<TOKENS / 16, 256, exp_lds, stream>>>(
      xbf, we, exnorm_w, exnorm_b, head_w, (float*)d_out);
}
